// SelfAttentionBlock_20366734917828
// MI455X (gfx1250) — compile-verified, matrix operand order checked by reading, not run on hardware
//
#include <hip/hip_runtime.h>

typedef __attribute__((ext_vector_type(16))) _Float16 v16h;
typedef __attribute__((ext_vector_type(8)))  float    v8f;
typedef __attribute__((ext_vector_type(4)))  int      v4i;

#define BB 8
#define CC 128
#define SS 1024
#define HH 8
#define GG 32
#define DD 128
#define N3 3072
#define EPSF 1e-5f

#if defined(__gfx1250__) && __has_builtin(__builtin_amdgcn_global_load_async_to_lds_b128)
#define HAS_ASYNC_LDS 1
#else
#define HAS_ASYNC_LDS 0
#endif

// Stage 16B global -> LDS via GLOBAL_LOAD_ASYNC_TO_LDS_B128 (ASYNCcnt-tracked,
// no VGPR round trip); fallback is load+ds_store.
__device__ __forceinline__ void stage_b128(const _Float16* g, _Float16* l) {
#if HAS_ASYNC_LDS
  __builtin_amdgcn_global_load_async_to_lds_b128(
      (v4i __attribute__((address_space(1)))*)(uintptr_t)g,
      (v4i __attribute__((address_space(3)))*)(uint32_t)(uintptr_t)l,
      0, 0);
#else
  *(uint4*)l = *(const uint4*)g;
#endif
}

#if HAS_ASYNC_LDS
#if __has_builtin(__builtin_amdgcn_s_wait_asynccnt)
#define STAGE_WAIT(n) __builtin_amdgcn_s_wait_asynccnt(n)
#else
#define STAGE_WAIT(n) asm volatile("s_wait_asynccnt %0" ::"i"(n) : "memory")
#endif
#else
#define STAGE_WAIT(n) ((void)0)
#endif

// A lane's 16 f16 fragment elements are two contiguous 8-element K-runs:
// K = base + [0..7] and K = base + 16 + [0..7], base = (lane&16)>>1  (ISA 7.12.2)
__device__ __forceinline__ v16h load_frag_h(const _Float16* p) {
  v16h r;
  ((uint4*)&r)[0] = ((const uint4*)p)[0];        // elements 0..7
  ((uint4*)&r)[1] = ((const uint4*)(p + 16))[0]; // elements 8..15
  return r;
}

// ---------------- Kernel 1: GroupNorm + transpose to [b, s, c] f16 ----------------
__global__ void gn_transpose_kernel(const float* __restrict__ x,
                                    const float* __restrict__ nw,
                                    const float* __restrict__ nb,
                                    _Float16* __restrict__ units_t) {
  __shared__ float s_sum[256];
  __shared__ float s_sq[256];
  int b = blockIdx.x >> 5;
  int g = blockIdx.x & 31;
  int t = threadIdx.x;
  const float* xp = x + ((size_t)(b * CC + g * 4)) * SS;  // 4 channels x 1024
  float vals[16];
  float sum = 0.f, sq = 0.f;
#pragma unroll
  for (int r = 0; r < 16; ++r) {
    float v = xp[t + r * 256];
    vals[r] = v;
    sum += v; sq += v * v;
  }
  s_sum[t] = sum; s_sq[t] = sq;
  __syncthreads();
  for (int off = 128; off > 0; off >>= 1) {
    if (t < off) { s_sum[t] += s_sum[t + off]; s_sq[t] += s_sq[t + off]; }
    __syncthreads();
  }
  float mean = s_sum[0] * (1.0f / 4096.0f);
  float var  = s_sq[0] * (1.0f / 4096.0f) - mean * mean;
  float inv  = __frsqrt_rn(var + EPSF);
#pragma unroll
  for (int r = 0; r < 16; ++r) {
    int f = t + r * 256;      // 0..4095
    int cl = f >> 10;         // channel within group
    int sp = f & 1023;        // spatial
    int ch = g * 4 + cl;
    float u = (vals[r] - mean) * inv * nw[ch] + nb[ch];
    units_t[((size_t)b * SS + sp) * CC + ch] = (_Float16)u;
  }
}

// ---------------- f32 [K][N] -> f16 transposed [N][K] ----------------
__global__ void cvt_transpose_kernel(const float* __restrict__ src,
                                     _Float16* __restrict__ dst,
                                     int K, int N) {
  int idx = blockIdx.x * 256 + threadIdx.x;
  if (idx >= K * N) return;
  int n = idx / K;
  int k = idx - n * K;
  dst[idx] = (_Float16)src[(size_t)k * N + n];
}

// ---------------- Kernel 2: QKV GEMM (M=1024,N=3072,K=128 per batch) -------------
#define QLDA 136
#define QLDB 136
__global__ void qkv_gemm_kernel(const _Float16* __restrict__ units_t,  // [B][S][C]
                                const _Float16* __restrict__ wi_t,     // [3072][128]
                                _Float16* __restrict__ qh,             // [B][H][S][D]
                                _Float16* __restrict__ kh,             // [B][H][S][D]
                                _Float16* __restrict__ vT) {           // [B][H][D][S]
  extern __shared__ char smem[];
  _Float16* As = (_Float16*)smem;                        // [128][QLDA] rows=m, K-contig
  _Float16* Bs = (_Float16*)(smem + 128 * QLDA * 2);     // [64][QLDB]  rows=n, K-contig
  int m0 = blockIdx.x * 128;
  int n0 = blockIdx.y * 64;
  int b  = blockIdx.z;
  int t = threadIdx.x;
  int lane = t & 31, wid = t >> 5;
  int kbase = (lane & 16) >> 1;
  {
    int row = t >> 1, koff = (t & 1) * 64;
    const _Float16* src = units_t + ((size_t)b * SS + m0 + row) * CC + koff;
    _Float16* dst = As + row * QLDA + koff;
#pragma unroll
    for (int u = 0; u < 8; ++u)
      stage_b128(src + u * 8, dst + u * 8);
  }
  {
    int row = t >> 2, koff = (t & 3) * 32;
    const _Float16* src = wi_t + (size_t)(n0 + row) * CC + koff;
    _Float16* dst = Bs + row * QLDB + koff;
#pragma unroll
    for (int u = 0; u < 4; ++u)
      stage_b128(src + u * 8, dst + u * 8);
  }
  STAGE_WAIT(0);
  __syncthreads();

  int wm = (wid & 3) * 32;
  int wn = (wid >> 2) * 32;
  v8f acc[2][2] = {};
#pragma unroll
  for (int kc = 0; kc < 4; ++kc) {
    v16h a[2], bf[2];
#pragma unroll
    for (int mt = 0; mt < 2; ++mt)
      a[mt] = load_frag_h(As + (wm + mt * 16 + (lane & 15)) * QLDA + kc * 32 + kbase);
#pragma unroll
    for (int nt = 0; nt < 2; ++nt)
      bf[nt] = load_frag_h(Bs + (wn + nt * 16 + (lane & 15)) * QLDB + kc * 32 + kbase);
#pragma unroll
    for (int mt = 0; mt < 2; ++mt)
#pragma unroll
      for (int nt = 0; nt < 2; ++nt)
        acc[mt][nt] = __builtin_amdgcn_wmma_f32_16x16x32_f16(
            false, a[mt], false, bf[nt], (short)0, acc[mt][nt], false, false);
  }
#pragma unroll
  for (int mt = 0; mt < 2; ++mt)
#pragma unroll
    for (int nt = 0; nt < 2; ++nt)
#pragma unroll
      for (int j = 0; j < 8; ++j) {
        int m = m0 + wm + mt * 16 + j + ((lane & 16) >> 1);  // s index
        int n = n0 + wn + nt * 16 + (lane & 15);
        int which = n >> 10;
        int rem = n & 1023;
        int head = rem >> 7;
        int d = rem & 127;
        _Float16 val = (_Float16)acc[mt][nt][j];
        size_t bh = (size_t)b * HH + head;
        if (which == 0)      qh[(bh * SS + m) * DD + d] = val;
        else if (which == 1) kh[(bh * SS + m) * DD + d] = val;
        else                 vT[(bh * DD + d) * SS + m] = val;  // V stored transposed
      }
}

// ---------------- Kernel 3: attention, one block per (b,h,16 query rows) ---------
// Double-buffered async K/V streaming; V tile 0 DMA hidden behind softmax.
#define ALDQ 136
#define ALDK 136
#define SCLD 1028
#define PLD  1032
__global__ void attention_kernel(const _Float16* __restrict__ qh,
                                 const _Float16* __restrict__ kh,
                                 const _Float16* __restrict__ vT,
                                 _Float16* __restrict__ oc) {  // [B][S][H*C]
  extern __shared__ char smem[];
  _Float16* Qt  = (_Float16*)smem;                                   // [16][ALDQ]
  float*    Sc  = (float*)(smem + 16 * ALDQ * 2);                    // [16][SCLD] f32
  _Float16* Pt  = (_Float16*)(smem + 16 * ALDQ * 2 + 16 * SCLD * 4); // [16][PLD] f16
  _Float16* KV0 = (_Float16*)(smem + 16 * ALDQ * 2 + 16 * SCLD * 4 + 16 * PLD * 2);
  _Float16* KV1 = KV0 + 128 * ALDK;
  float* rowsum = (float*)(smem + 16 * ALDQ * 2 + 16 * SCLD * 4 + 16 * PLD * 2 +
                           2 * 128 * ALDK * 2);

  int m0 = blockIdx.x * 16;
  int h  = blockIdx.y;
  int b  = blockIdx.z;
  int t = threadIdx.x, lane = t & 31, wid = t >> 5;
  int kbase = (lane & 16) >> 1;
  size_t bh = (size_t)b * HH + h;
  const _Float16* Qg = qh + bh * SS * DD;
  const _Float16* Kg = kh + bh * SS * DD;
  const _Float16* Vg = vT + bh * DD * SS;  // [d][s]

  int row = t >> 1, coff = (t & 1) * 64;   // staging split: 2 threads per row

  {  // prologue: Q tile + K tile 0
    int qrow = t >> 4, qcol = (t & 15) * 8;
    stage_b128(Qg + (size_t)(m0 + qrow) * DD + qcol, Qt + qrow * ALDQ + qcol);
    const _Float16* src = Kg + (size_t)row * DD + coff;
    _Float16* dst = KV0 + row * ALDK + coff;
#pragma unroll
    for (int u = 0; u < 8; ++u)
      stage_b128(src + u * 8, dst + u * 8);
  }
  STAGE_WAIT(0);
  __syncthreads();
  v16h aq[4];
#pragma unroll
  for (int kc = 0; kc < 4; ++kc)
    aq[kc] = load_frag_h(Qt + (lane & 15) * ALDQ + kc * 32 + kbase);

  const float scale = 0.08838834764831845f;  // 1/sqrt(128)
  // ---- phase 1: S = Q K^T * scale, double-buffered K stream ----
  for (int kb = 0; kb < 8; ++kb) {
    _Float16* cur = (kb & 1) ? KV1 : KV0;
    if (kb < 7) {
      _Float16* nxt = (kb & 1) ? KV0 : KV1;
      const _Float16* src = Kg + (size_t)((kb + 1) * 128 + row) * DD + coff;
      _Float16* dst = nxt + row * ALDK + coff;
#pragma unroll
      for (int u = 0; u < 8; ++u)
        stage_b128(src + u * 8, dst + u * 8);
      STAGE_WAIT(8);   // asyncs complete in order: tile kb done
    } else {
      STAGE_WAIT(0);
    }
    __syncthreads();
    v8f acc = {};
    int keyl = wid * 16;
#pragma unroll
    for (int kc = 0; kc < 4; ++kc) {
      v16h bf = load_frag_h(cur + (keyl + (lane & 15)) * ALDK + kc * 32 + kbase);
      acc = __builtin_amdgcn_wmma_f32_16x16x32_f16(
          false, aq[kc], false, bf, (short)0, acc, false, false);
    }
#pragma unroll
    for (int j = 0; j < 8; ++j) {
      int m = j + ((lane & 16) >> 1);
      int key = kb * 128 + keyl + (lane & 15);
      Sc[m * SCLD + key] = acc[j] * scale;
    }
    __syncthreads();   // protect buffer reuse by next iteration's stage
  }
  {  // issue V tile 0 DMA; it runs concurrently with softmax below
    const _Float16* src = Vg + (size_t)row * SS + coff;
    _Float16* dst = KV0 + row * ALDK + coff;
#pragma unroll
    for (int u = 0; u < 8; ++u)
      stage_b128(src + u * 8, dst + u * 8);
  }
  // ---- phase 2: softmax; write P as f16 (16 threads per row) ----
  {
    int srow = t >> 4, cidx = t & 15;
    float mx = -1e30f;
#pragma unroll 8
    for (int u = 0; u < 64; ++u)
      mx = fmaxf(mx, Sc[srow * SCLD + cidx + u * 16]);
#pragma unroll
    for (int msk = 8; msk >= 1; msk >>= 1)
      mx = fmaxf(mx, __shfl_xor(mx, msk, 32));
    float sum = 0.f;
#pragma unroll 8
    for (int u = 0; u < 64; ++u) {
      int col = cidx + u * 16;
      float p = __expf(Sc[srow * SCLD + col] - mx);
      Pt[srow * PLD + col] = (_Float16)p;
      sum += p;
    }
#pragma unroll
    for (int msk = 8; msk >= 1; msk >>= 1)
      sum += __shfl_xor(sum, msk, 32);
    if (cidx == 0) rowsum[srow] = sum;
  }
  STAGE_WAIT(0);
  __syncthreads();   // softmax results + V0 visible to all waves
  // ---- phase 3: O = P V, double-buffered V stream; wave owns 16 d columns ----
  v8f accO = {};
  for (int kb = 0; kb < 8; ++kb) {
    _Float16* cur = (kb & 1) ? KV1 : KV0;
    if (kb < 7) {
      _Float16* nxt = (kb & 1) ? KV0 : KV1;
      const _Float16* src = Vg + (size_t)row * SS + (kb + 1) * 128 + coff;
      _Float16* dst = nxt + row * ALDK + coff;
#pragma unroll
      for (int u = 0; u < 8; ++u)
        stage_b128(src + u * 8, dst + u * 8);
      STAGE_WAIT(8);
    } else {
      STAGE_WAIT(0);
    }
    __syncthreads();
#pragma unroll
    for (int kc = 0; kc < 4; ++kc) {
      v16h ap = load_frag_h(Pt + (lane & 15) * PLD + kb * 128 + kc * 32 + kbase);
      v16h bf = load_frag_h(cur + (wid * 16 + (lane & 15)) * ALDK + kc * 32 + kbase);
      accO = __builtin_amdgcn_wmma_f32_16x16x32_f16(
          false, ap, false, bf, (short)0, accO, false, false);
    }
    __syncthreads();
  }
#pragma unroll
  for (int j = 0; j < 8; ++j) {
    int m = j + ((lane & 16) >> 1);
    int d = wid * 16 + (lane & 15);
    float o = accO[j] / rowsum[m];
    oc[((size_t)b * SS + m0 + m) * (HH * CC) + h * CC + d] = (_Float16)o;
  }
}

// ---------------- Kernel 4: out-proj GEMM (8192x128x1024) + residual -------------
// Double-buffered async K-chunk streaming.
#define OLDA 40
#define OLDB 40
__global__ void out_proj_kernel(const _Float16* __restrict__ oc,    // [B*S][1024]
                                const _Float16* __restrict__ wo_t,  // [128][1024]
                                const float* __restrict__ x,
                                float* __restrict__ out) {
  extern __shared__ char smem[];
  _Float16* A0 = (_Float16*)smem;
  _Float16* B0 = (_Float16*)(smem + 128 * OLDA * 2);
  _Float16* A1 = (_Float16*)(smem + 2 * 128 * OLDA * 2);
  _Float16* B1 = (_Float16*)(smem + 3 * 128 * OLDA * 2);
  int r0 = blockIdx.x * 128;
  int t = threadIdx.x, lane = t & 31, wid = t >> 5;
  int kbase = (lane & 16) >> 1;
  int wm = (wid & 3) * 32;
  int wn = (wid >> 2) * 64;
  int row = t >> 1, koff = (t & 1) * 16;

  {  // prologue: stage chunk 0
    const _Float16* srcA = oc + (size_t)(r0 + row) * 1024 + koff;
    stage_b128(srcA, A0 + row * OLDA + koff);
    stage_b128(srcA + 8, A0 + row * OLDA + koff + 8);
    const _Float16* srcB = wo_t + (size_t)row * 1024 + koff;
    stage_b128(srcB, B0 + row * OLDB + koff);
    stage_b128(srcB + 8, B0 + row * OLDB + koff + 8);
  }
  v8f acc[2][4] = {};
  for (int kk = 0; kk < 32; ++kk) {
    _Float16* Ac = (kk & 1) ? A1 : A0;
    _Float16* Bc = (kk & 1) ? B1 : B0;
    if (kk < 31) {
      _Float16* An = (kk & 1) ? A0 : A1;
      _Float16* Bn = (kk & 1) ? B0 : B1;
      const _Float16* srcA = oc + (size_t)(r0 + row) * 1024 + (kk + 1) * 32 + koff;
      stage_b128(srcA, An + row * OLDA + koff);
      stage_b128(srcA + 8, An + row * OLDA + koff + 8);
      const _Float16* srcB = wo_t + (size_t)row * 1024 + (kk + 1) * 32 + koff;
      stage_b128(srcB, Bn + row * OLDB + koff);
      stage_b128(srcB + 8, Bn + row * OLDB + koff + 8);
      STAGE_WAIT(4);   // in-order completion: chunk kk landed
    } else {
      STAGE_WAIT(0);
    }
    __syncthreads();
    v16h a[2], bf[4];
#pragma unroll
    for (int mt = 0; mt < 2; ++mt)
      a[mt] = load_frag_h(Ac + (wm + mt * 16 + (lane & 15)) * OLDA + kbase);
#pragma unroll
    for (int nt = 0; nt < 4; ++nt)
      bf[nt] = load_frag_h(Bc + (wn + nt * 16 + (lane & 15)) * OLDB + kbase);
#pragma unroll
    for (int mt = 0; mt < 2; ++mt)
#pragma unroll
      for (int nt = 0; nt < 4; ++nt)
        acc[mt][nt] = __builtin_amdgcn_wmma_f32_16x16x32_f16(
            false, a[mt], false, bf[nt], (short)0, acc[mt][nt], false, false);
    __syncthreads();   // protect buffer reuse by next iteration's stage
  }
#pragma unroll
  for (int mt = 0; mt < 2; ++mt)
#pragma unroll
    for (int nt = 0; nt < 4; ++nt)
#pragma unroll
      for (int j = 0; j < 8; ++j) {
        int r = r0 + wm + mt * 16 + j + ((lane & 16) >> 1);
        int n = wn + nt * 16 + (lane & 15);
        size_t p = (size_t)r * 128 + n;
        out[p] = x[p] + acc[mt][nt][j];  // raw-reshape residual: flat add
      }
}

extern "C" void kernel_launch(void* const* d_in, const int* in_sizes, int n_in,
                              void* d_out, int out_size, void* d_ws, size_t ws_size,
                              hipStream_t stream) {
  const float* x  = (const float*)d_in[0];
  const float* nw = (const float*)d_in[1];
  const float* nb = (const float*)d_in[2];
  const float* wi = (const float*)d_in[3];
  const float* wo = (const float*)d_in[4];
  float* out = (float*)d_out;

  char* ws = (char*)d_ws;
  size_t off = 0;
  auto wsalloc = [&](size_t bytes) {
    char* p = ws + off;
    off = (off + bytes + 255) & ~(size_t)255;
    return p;
  };
  _Float16* units_t = (_Float16*)wsalloc((size_t)BB * SS * CC * 2);
  _Float16* wi_t    = (_Float16*)wsalloc((size_t)N3 * CC * 2);        // [3072][128]
  _Float16* wo_t    = (_Float16*)wsalloc((size_t)CC * HH * CC * 2);   // [128][1024]
  _Float16* qh      = (_Float16*)wsalloc((size_t)BB * HH * SS * DD * 2);
  _Float16* kh      = (_Float16*)wsalloc((size_t)BB * HH * SS * DD * 2);
  _Float16* vTw     = (_Float16*)wsalloc((size_t)BB * HH * DD * SS * 2);
  _Float16* ocn     = (_Float16*)wsalloc((size_t)BB * SS * HH * CC * 2);

  gn_transpose_kernel<<<dim3(BB * GG), dim3(256), 0, stream>>>(x, nw, nb, units_t);

  int nwi = CC * N3;
  cvt_transpose_kernel<<<dim3((nwi + 255) / 256), dim3(256), 0, stream>>>(
      wi, wi_t, CC, N3);          // wi [128][3072] -> wi_t [3072][128]
  int nwo = HH * CC * CC;
  cvt_transpose_kernel<<<dim3((nwo + 255) / 256), dim3(256), 0, stream>>>(
      wo, wo_t, HH * CC, CC);     // wo [1024][128] -> wo_t [128][1024]

  size_t qkv_smem = 128 * QLDA * 2 + 64 * QLDB * 2;
  qkv_gemm_kernel<<<dim3(SS / 128, N3 / 64, BB), dim3(256), qkv_smem, stream>>>(
      units_t, wi_t, qh, kh, vTw);

  size_t att_smem = 16 * ALDQ * 2 + 16 * SCLD * 4 + 16 * PLD * 2 +
                    2 * 128 * ALDK * 2 + 64;
  attention_kernel<<<dim3(SS / 16, HH, BB), dim3(256), att_smem, stream>>>(
      qh, kh, vTw, ocn);

  size_t op_smem = 4 * 128 * OLDA * 2;
  out_proj_kernel<<<dim3((BB * SS) / 128), dim3(256), op_smem, stream>>>(
      ocn, wo_t, x, out);
}
